// AUAttnProcessor_20134806684198
// MI455X (gfx1250) — matrix operand order repaired and checked by reading, not compiled, not run on hardware
//
#include <hip/hip_runtime.h>
#include <math.h>

typedef _Float16 v16h __attribute__((ext_vector_type(16)));
typedef _Float16 v8h  __attribute__((ext_vector_type(8)));
typedef float    v8f  __attribute__((ext_vector_type(8)));
typedef int      v4i  __attribute__((ext_vector_type(4)));

#define AS1 __attribute__((address_space(1)))
#define AS3 __attribute__((address_space(3)))

// --------------------------------------------------------------------------
// gfx1250 async global->LDS staging (ASYNCcnt). The builtin exists on this
// toolchain with signature (v4i AS1*, v4i AS3*, imm offset, imm cpol).
// --------------------------------------------------------------------------
#if __has_builtin(__builtin_amdgcn_global_load_async_to_lds_b128)
#define ASYNC_LDS 1
#endif

#if defined(ASYNC_LDS)
#if __has_builtin(__builtin_amdgcn_s_wait_asynccnt)
#define WAIT_ASYNC(n) __builtin_amdgcn_s_wait_asynccnt(n)
#else
#define WAIT_ASYNC(n) asm volatile("s_wait_asynccnt %0" ::"i"(n) : "memory")
#endif
#else
#define WAIT_ASYNC(n) ((void)0)
#endif

__device__ inline void lds_copy16(const _Float16* __restrict__ src,
                                  _Float16* __restrict__ dst) {
#if defined(ASYNC_LDS)
    _Float16* s = const_cast<_Float16*>(src);
    AS1 _Float16* sg = (AS1 _Float16*)s;     // generic -> global addrspace
    AS3 _Float16* dl = (AS3 _Float16*)dst;   // generic -> LDS addrspace
    __builtin_amdgcn_global_load_async_to_lds_b128(
        (AS1 v4i*)sg, (AS3 v4i*)dl, 0, 0);
#else
    *(v8h*)dst = *(const v8h*)src;
#endif
}

// ---------------------------------------------------------------------------
// DPP butterfly reductions across the 16-lane DPP row (no LDS, no waits).
// ---------------------------------------------------------------------------
template <int CTRL>
__device__ inline float dpp_mov_f32(float x) {
    int t = __builtin_amdgcn_update_dpp(0, __builtin_bit_cast(int, x),
                                        CTRL, 0xF, 0xF, true);
    return __builtin_bit_cast(float, t);
}
__device__ inline float red_max16(float x) {
    x = fmaxf(x, dpp_mov_f32<0xB1>(x));    // xor1: quad_perm(1,0,3,2)
    x = fmaxf(x, dpp_mov_f32<0x4E>(x));    // xor2: quad_perm(2,3,0,1)
    x = fmaxf(x, dpp_mov_f32<0x141>(x));   // ^7 : row_half_mirror
    x = fmaxf(x, dpp_mov_f32<0x140>(x));   // ^15: row_mirror
    return x;
}
__device__ inline float red_sum16(float x) {
    x += dpp_mov_f32<0xB1>(x);
    x += dpp_mov_f32<0x4E>(x);
    x += dpp_mov_f32<0x141>(x);
    x += dpp_mov_f32<0x140>(x);
    return x;
}

// ---------------------------------------------------------------------------
// casts
// ---------------------------------------------------------------------------
__global__ __launch_bounds__(256)
void cast_f32_f16(const float* __restrict__ in, _Float16* __restrict__ out, int n) {
    int i = blockIdx.x * 256 + threadIdx.x;
    if (i < n) out[i] = (_Float16)in[i];
}

// W[K,N] f32 -> Wt[N,K] f16
__global__ __launch_bounds__(256)
void cast_transpose(const float* __restrict__ in, _Float16* __restrict__ out,
                    int K, int N) {
    int i = blockIdx.x * 256 + threadIdx.x;
    if (i < K * N) {
        int k = i / N, n = i % N;
        out[(size_t)n * K + k] = (_Float16)in[i];
    }
}

// V [B,S,D] f16 -> Vt [B,D,S] f16
__global__ __launch_bounds__(256)
void transpose_v(const _Float16* __restrict__ in, _Float16* __restrict__ out,
                 int S, int D, int n) {
    int i = blockIdx.x * 256 + threadIdx.x;
    if (i >= n) return;
    int d = i % D;
    int t = i / D;
    int s = t % S;
    int b = t / S;
    out[((size_t)b * D + d) * S + s] = in[i];
}

// ---------------------------------------------------------------------------
// WMMA GEMM: C[M,N] = A[M,K] @ B[K,N] with B supplied TRANSPOSED (Bt[N,K]).
// One wave per 16x64 tile (4 accumulators share one A-fragment).
// EPI: 0 f32 | 1 f16 | 2 silu(x+bias)->f16 | 3 sigmoid(x+bias)->f32
//      4 x+bias+residual->f32
// ---------------------------------------------------------------------------
template <int EPI>
__global__ __launch_bounds__(32)
void gemm_wmma(const _Float16* __restrict__ A, int lda,
               const _Float16* __restrict__ Bt, int ldb /* = K */,
               int K,
               float* __restrict__ Cf, _Float16* __restrict__ Ch, int ldc,
               const float* __restrict__ bias,
               const float* __restrict__ residual) {
    const int lane = threadIdx.x;
    const int hl   = lane >> 4;
    const int l15  = lane & 15;
    const int tm   = blockIdx.y;
    const int tn   = blockIdx.x;

    const int arow = tm * 16 + l15;

    v8f acc[4];
#pragma unroll
    for (int t = 0; t < 4; ++t) acc[t] = (v8f){};

    for (int k0 = 0; k0 < K; k0 += 32) {
        const _Float16* ap = A + (size_t)arow * lda + k0;
        v8h a0 = *(const v8h*)(ap + hl * 8);
        v8h a1 = *(const v8h*)(ap + 16 + hl * 8);
        v16h af;
#pragma unroll
        for (int e = 0; e < 8; ++e) { af[e] = a0[e]; af[e + 8] = a1[e]; }

#pragma unroll
        for (int t = 0; t < 4; ++t) {
            const int bcol = tn * 64 + t * 16 + l15;
            const _Float16* bp = Bt + (size_t)bcol * ldb + k0 + hl * 8;
            v8h b0 = *(const v8h*)bp;
            v8h b1 = *(const v8h*)(bp + 16);
            v16h bf;
#pragma unroll
            for (int e = 0; e < 8; ++e) { bf[e] = b0[e]; bf[e + 8] = b1[e]; }
            acc[t] = __builtin_amdgcn_wmma_f32_16x16x32_f16(
                false, af, false, bf, (short)0, acc[t], false, false);
        }
    }

#pragma unroll
    for (int t = 0; t < 4; ++t) {
#pragma unroll
        for (int r = 0; r < 8; ++r) {
            const int orow = tm * 16 + hl * 8 + r;
            const int ocol = tn * 64 + t * 16 + l15;
            float x = acc[t][r];
            if (EPI >= 2) x += bias[ocol];
            if (EPI == 0) {
                Cf[(size_t)orow * ldc + ocol] = x;
            } else if (EPI == 1) {
                Ch[(size_t)orow * ldc + ocol] = (_Float16)x;
            } else if (EPI == 2) {
                float s = x / (1.0f + __expf(-x));   // SiLU
                Ch[(size_t)orow * ldc + ocol] = (_Float16)s;
            } else if (EPI == 3) {
                Cf[(size_t)orow * ldc + ocol] = 1.0f / (1.0f + __expf(-x));
            } else {                                  // 4
                Cf[(size_t)orow * ldc + ocol] = x + residual[(size_t)orow * ldc + ocol];
            }
        }
    }
}

// ---------------------------------------------------------------------------
// Flash attention, 4-wave workgroup per (b, h, 64 Q rows).
// K chunk (32 keys x 96 padded dims) and Vt chunk (80 dims x 32 keys) are
// staged cooperatively into LDS (async, double-buffered) and shared by all
// 4 waves; each wave owns one 16-row Q tile.
// ---------------------------------------------------------------------------
#define KW 96   // padded K-row width in LDS (80 data + 16 zeros)

__global__ __launch_bounds__(128)
void flash_attn(const _Float16* __restrict__ Q, const _Float16* __restrict__ K,
                const _Float16* __restrict__ Vt, float* __restrict__ O,
                int S, int D, int dh, int H, float scale) {
    __shared__ alignas(16) _Float16 Kb[2][32 * KW];
    __shared__ alignas(16) _Float16 Vb[2][80 * 32];
    __shared__ alignas(16) _Float16 Pw[4][16 * 32];

    const int tid  = threadIdx.x;
    const int wave = tid >> 5;
    const int lane = tid & 31;
    const int hl   = lane >> 4;
    const int l15  = lane & 15;
    const int bh   = blockIdx.y;
    const int bb   = bh / H, hh = bh % H;
    const int qt   = blockIdx.x * 4 + wave;          // this wave's 16-row tile
    const size_t base  = (size_t)bb * S * D + (size_t)hh * dh;   // Q,K,O
    const size_t vbase = ((size_t)bb * D + (size_t)hh * dh) * S; // Vt
    const _Float16* Kg  = K  + base;
    const _Float16* Vtg = Vt + vbase;

    // zero the d-pad columns [80,96) of both K buffers (written once)
    for (int i = tid; i < 2 * 32 * 16; i += 128) {
        int buf = i >> 9;            // /512
        int rem = i & 511;
        int row = rem >> 4, c = rem & 15;
        Kb[buf][row * KW + 80 + c] = (_Float16)0.f;
    }

    // stage one chunk: 320 K segments (32 rows x 10) + 320 Vt segments
    // (80 cols x 4) = 640 x 16B; 128 threads x 5 rounds, exact.
    auto stage = [&](int buf, int kc) {
#pragma unroll
        for (int r = 0; r < 5; ++r) {
            int seg = r * 128 + tid;
            const _Float16* src;
            _Float16* dst;
            if (seg < 320) {
                int row = seg / 10, c = seg % 10;
                src = Kg + (size_t)(kc + row) * D + c * 8;
                dst = &Kb[buf][row * KW + c * 8];
            } else {
                int s2 = seg - 320;
                int col = s2 >> 2, c = s2 & 3;
                src = Vtg + (size_t)col * S + kc + c * 8;
                dst = &Vb[buf][col * 32 + c * 8];
            }
            lds_copy16(src, dst);
        }
    };

    // --- preload Q A-fragments (k-chunks 0,32,64), scale folded in ---
    v16h aq[3];
    const _Float16* qrow = Q + base + (size_t)(qt * 16 + l15) * D;
#pragma unroll
    for (int kk = 0; kk < 3; ++kk) {
        const _Float16* qp = qrow + kk * 32;
        v8h q0 = *(const v8h*)(qp + hl * 8);
#pragma unroll
        for (int e = 0; e < 8; ++e)
            aq[kk][e] = (_Float16)((float)q0[e] * scale);
        if (kk < 2) {
            v8h q1 = *(const v8h*)(qp + 16 + hl * 8);
#pragma unroll
            for (int e = 0; e < 8; ++e)
                aq[kk][e + 8] = (_Float16)((float)q1[e] * scale);
        } else {
#pragma unroll
            for (int e = 0; e < 8; ++e) aq[kk][e + 8] = (_Float16)0.f;
        }
    }

    v8f acc[5];
    float m[8], l[8];
#pragma unroll
    for (int nb = 0; nb < 5; ++nb) acc[nb] = (v8f){};
#pragma unroll
    for (int r = 0; r < 8; ++r) { m[r] = -1e30f; l[r] = 0.f; }

    const int NC = S / 32;
    stage(0, 0);
    for (int c = 0; c < NC; ++c) {
        const int buf = c & 1;
        if (c + 1 < NC) {
            stage(buf ^ 1, (c + 1) * 32);
            WAIT_ASYNC(5);           // previous chunk complete, next in flight
        } else {
            WAIT_ASYNC(0);
        }
        __syncthreads();             // chunk c visible to all 4 waves

        const _Float16* kb = &Kb[buf][0];
        const _Float16* vb = &Vb[buf][0];

        // ---- logits: s0 = keys 0..15 of chunk, s1 = keys 16..31 ----
        v8f s0 = {}, s1 = {};
#pragma unroll
        for (int kk = 0; kk < 3; ++kk) {
            const _Float16* kp0 = kb + l15 * KW + kk * 32 + hl * 8;        // [kd,kd+8) and [kd+16,kd+24)
            const _Float16* kp1 = kb + (16 + l15) * KW + kk * 32 + hl * 8; // pad gives zeros
            v8h x0 = *(const v8h*)kp0, x1 = *(const v8h*)(kp0 + 16);
            v8h y0 = *(const v8h*)kp1, y1 = *(const v8h*)(kp1 + 16);
            v16h bk0, bk1;
#pragma unroll
            for (int e = 0; e < 8; ++e) {
                bk0[e] = x0[e]; bk0[e + 8] = x1[e];
                bk1[e] = y0[e]; bk1[e + 8] = y1[e];
            }
            s0 = __builtin_amdgcn_wmma_f32_16x16x32_f16(false, aq[kk], false, bk0, (short)0, s0, false, false);
            s1 = __builtin_amdgcn_wmma_f32_16x16x32_f16(false, aq[kk], false, bk1, (short)0, s1, false, false);
        }

        // ---- online softmax update (DPP reductions) ----
        _Float16* pb = &Pw[wave][0];
#pragma unroll
        for (int r = 0; r < 8; ++r) {
            float cm = red_max16(fmaxf(s0[r], s1[r]));
            float nm = fmaxf(m[r], cm);
            float al = __expf(m[r] - nm);
            float p0 = __expf(s0[r] - nm);
            float p1 = __expf(s1[r] - nm);
            float rs = red_sum16(p0 + p1);
            l[r] = l[r] * al + rs;
            m[r] = nm;
#pragma unroll
            for (int nb = 0; nb < 5; ++nb) acc[nb][r] *= al;
            pb[(hl * 8 + r) * 32 + l15]      = (_Float16)p0;
            pb[(hl * 8 + r) * 32 + 16 + l15] = (_Float16)p1;
        }
        asm volatile("s_wait_dscnt 0x0" ::: "memory");  // wave-private P tile

        v16h ap;
        {
            v8h p0 = *(const v8h*)&pb[l15 * 32 + hl * 8];
            v8h p1 = *(const v8h*)&pb[l15 * 32 + 16 + hl * 8];
#pragma unroll
            for (int e = 0; e < 8; ++e) { ap[e] = p0[e]; ap[e + 8] = p1[e]; }
        }

        // ---- PV: acc[nb] += P @ V ----
#pragma unroll
        for (int nb = 0; nb < 5; ++nb) {
            const _Float16* vp = vb + (nb * 16 + l15) * 32 + hl * 8;
            v8h x0 = *(const v8h*)vp, x1 = *(const v8h*)(vp + 16);
            v16h bv;
#pragma unroll
            for (int e = 0; e < 8; ++e) { bv[e] = x0[e]; bv[e + 8] = x1[e]; }
            acc[nb] = __builtin_amdgcn_wmma_f32_16x16x32_f16(false, ap, false, bv, (short)0, acc[nb], false, false);
        }
        __syncthreads();             // all waves done with chunk c buffers
    }

    // ---- normalize and write ----
#pragma unroll
    for (int r = 0; r < 8; ++r) {
        float inv = 1.0f / l[r];
        int orow  = qt * 16 + hl * 8 + r;
#pragma unroll
        for (int nb = 0; nb < 5; ++nb)
            O[base + (size_t)orow * D + nb * 16 + l15] = acc[nb][r] * inv;
    }
}

// ---------------------------------------------------------------------------
// AU cross-attention (A=16 keys): scalar, one thread per (b, s, h).
// ---------------------------------------------------------------------------
__global__ __launch_bounds__(128)
void au_attn(const _Float16* __restrict__ Q, const float* __restrict__ AK,
             const float* __restrict__ AV, const float* __restrict__ temp,
             float* __restrict__ AUH, _Float16* __restrict__ AUHh,
             int BS, int S, int D, int dh, int H, float scale) {
    const int idx = blockIdx.x * 128 + threadIdx.x;
    if (idx >= BS * H) return;
    const int h  = idx % H;
    const int bs = idx / H;
    const int b  = bs / S;
    const float alpha = scale * temp[0];

    const _Float16* q  = Q  + (size_t)bs * D + h * dh;
    const float*    ak = AK + (size_t)b * 16 * D + h * dh;
    const float*    av = AV + (size_t)b * 16 * D + h * dh;

    float logits[16];
    float mx = -1e30f;
#pragma unroll
    for (int a = 0; a < 16; ++a) {
        float dot = 0.f;
        for (int j = 0; j < dh; ++j)
            dot += (float)q[j] * ak[(size_t)a * D + j];
        dot *= alpha;
        logits[a] = dot;
        mx = fmaxf(mx, dot);
    }
    float denom = 0.f;
#pragma unroll
    for (int a = 0; a < 16; ++a) { logits[a] = __expf(logits[a] - mx); denom += logits[a]; }
    const float inv = 1.f / denom;

    for (int j = 0; j < dh; ++j) {
        float o = 0.f;
#pragma unroll
        for (int a = 0; a < 16; ++a) o += logits[a] * av[(size_t)a * D + j];
        o *= inv;
        AUH [(size_t)bs * D + h * dh + j] = o;
        AUHh[(size_t)bs * D + h * dh + j] = (_Float16)o;
    }
}

// ---------------------------------------------------------------------------
// fused = hs + gate * au_hs   (f16 out for the final GEMM)
// ---------------------------------------------------------------------------
__global__ __launch_bounds__(256)
void fuse_gate(const float* __restrict__ hs, const float* __restrict__ gate,
               const float* __restrict__ auh, _Float16* __restrict__ out, int n) {
    int i = blockIdx.x * 256 + threadIdx.x;
    if (i < n) out[i] = (_Float16)(hs[i] + gate[i] * auh[i]);
}

// ---------------------------------------------------------------------------
// host launcher
// ---------------------------------------------------------------------------
extern "C" void kernel_launch(void* const* d_in, const int* in_sizes, int n_in,
                              void* d_out, int out_size, void* d_ws, size_t ws_size,
                              hipStream_t stream) {
    (void)in_sizes; (void)n_in; (void)out_size; (void)ws_size;
    const int B = 4, S = 2048, D = 640, C = 768, H = 8, dh = 80;
    const int M = B * S;                 // 8192 rows
    const int BA = B * 16;               // 64 au rows
    const float scale = 0.11180339887498949f; // 1/sqrt(80)

    const float* hs   = (const float*)d_in[0];
    const float* aue  = (const float*)d_in[1];
    const float* wq   = (const float*)d_in[2];
    const float* wk   = (const float*)d_in[3];
    const float* wv   = (const float*)d_in[4];
    const float* wak  = (const float*)d_in[5];
    const float* wav  = (const float*)d_in[6];
    const float* wg1  = (const float*)d_in[7];
    const float* bg1  = (const float*)d_in[8];
    const float* wg2  = (const float*)d_in[9];
    const float* bg2  = (const float*)d_in[10];
    const float* wout = (const float*)d_in[11];
    const float* bout = (const float*)d_in[12];
    const float* temp = (const float*)d_in[13];
    float* out = (float*)d_out;

    // workspace carve-up
    char* p = (char*)d_ws;
    auto alloc = [&](size_t bytes) -> void* {
        void* r = (void*)p; p += (bytes + 255) & ~(size_t)255; return r;
    };
    _Float16* hsh   = (_Float16*)alloc((size_t)M * D * 2);
    _Float16* aueh  = (_Float16*)alloc((size_t)BA * C * 2);
    _Float16* qh    = (_Float16*)alloc((size_t)M * D * 2);
    _Float16* kh    = (_Float16*)alloc((size_t)M * D * 2);
    _Float16* vh    = (_Float16*)alloc((size_t)M * D * 2);
    _Float16* vth   = (_Float16*)alloc((size_t)M * D * 2);   // V^T [B,D,S]
    _Float16* wqt   = (_Float16*)alloc((size_t)D * D * 2);   // transposed weights [N,K]
    _Float16* wkt   = (_Float16*)alloc((size_t)D * D * 2);
    _Float16* wvt   = (_Float16*)alloc((size_t)D * D * 2);
    _Float16* wakt  = (_Float16*)alloc((size_t)C * D * 2);
    _Float16* wavt  = (_Float16*)alloc((size_t)C * D * 2);
    _Float16* wg1t  = (_Float16*)alloc((size_t)D * (D / 2) * 2);
    _Float16* wg2t  = (_Float16*)alloc((size_t)(D / 2) * D * 2);
    _Float16* woutt = (_Float16*)alloc((size_t)D * D * 2);
    float*    aukf  = (float*)   alloc((size_t)BA * D * 4);
    float*    auvf  = (float*)   alloc((size_t)BA * D * 4);
    float*    attnO = (float*)   alloc((size_t)M * D * 4);
    float*    auhf  = (float*)   alloc((size_t)M * D * 4);
    _Float16* auhh  = (_Float16*)alloc((size_t)M * D * 2);
    _Float16* g1h   = (_Float16*)alloc((size_t)M * (D / 2) * 2);
    float*    gatef = (float*)   alloc((size_t)M * D * 4);
    _Float16* fusedh= (_Float16*)alloc((size_t)M * D * 2);

    auto cgrid = [](int n) { return dim3((n + 255) / 256); };

    // 1) casts (weights transposed on the fly: W[K,N] -> Wt[N,K])
    cast_f32_f16<<<cgrid(M * D), 256, 0, stream>>>(hs, hsh, M * D);
    cast_f32_f16<<<cgrid(BA * C), 256, 0, stream>>>(aue, aueh, BA * C);
    cast_transpose<<<cgrid(D * D), 256, 0, stream>>>(wq, wqt, D, D);
    cast_transpose<<<cgrid(D * D), 256, 0, stream>>>(wk, wkt, D, D);
    cast_transpose<<<cgrid(D * D), 256, 0, stream>>>(wv, wvt, D, D);
    cast_transpose<<<cgrid(C * D), 256, 0, stream>>>(wak, wakt, C, D);
    cast_transpose<<<cgrid(C * D), 256, 0, stream>>>(wav, wavt, C, D);
    cast_transpose<<<cgrid(D * D / 2), 256, 0, stream>>>(wg1, wg1t, D, D / 2);
    cast_transpose<<<cgrid(D * D / 2), 256, 0, stream>>>(wg2, wg2t, D / 2, D);
    cast_transpose<<<cgrid(D * D), 256, 0, stream>>>(wout, woutt, D, D);

    // 2) Q/K/V projections -> f16
    {
        dim3 g(D / 64, M / 16);
        gemm_wmma<1><<<g, 32, 0, stream>>>(hsh, D, wqt, D, D, nullptr, qh, D, nullptr, nullptr);
        gemm_wmma<1><<<g, 32, 0, stream>>>(hsh, D, wkt, D, D, nullptr, kh, D, nullptr, nullptr);
        gemm_wmma<1><<<g, 32, 0, stream>>>(hsh, D, wvt, D, D, nullptr, vh, D, nullptr, nullptr);
    }
    //    V -> V^T [B, D, S]
    transpose_v<<<cgrid(M * D), 256, 0, stream>>>(vh, vth, S, D, M * D);

    // 3) au_k / au_v projections -> f32
    {
        dim3 g(D / 64, BA / 16);
        gemm_wmma<0><<<g, 32, 0, stream>>>(aueh, C, wakt, C, C, aukf, nullptr, D, nullptr, nullptr);
        gemm_wmma<0><<<g, 32, 0, stream>>>(aueh, C, wavt, C, C, auvf, nullptr, D, nullptr, nullptr);
    }
    // 4) self-attention: 4-wave workgroups, LDS-shared K/V, async staging
    {
        dim3 g(S / 64, B * H);
        flash_attn<<<g, 128, 0, stream>>>(qh, kh, vth, attnO, S, D, dh, H, scale);
    }
    // 5) AU cross-attention (scalar, A=16)
    {
        int total = M * H;
        au_attn<<<dim3((total + 127) / 128), 128, 0, stream>>>(
            qh, aukf, auvf, temp, auhf, auhh, M, S, D, dh, H, scale);
    }
    // 6) gated MLP: g1 = silu(au_hs @ w_g1 + b_g1) -> f16
    {
        dim3 g((D / 2) / 64, M / 16);
        gemm_wmma<2><<<g, 32, 0, stream>>>(auhh, D, wg1t, D, D, nullptr, g1h, D / 2, bg1, nullptr);
    }
    //    gate = sigmoid(g1 @ w_g2 + b_g2) -> f32
    {
        dim3 g(D / 64, M / 16);
        gemm_wmma<3><<<g, 32, 0, stream>>>(g1h, D / 2, wg2t, D / 2, D / 2, gatef, nullptr, D, bg2, nullptr);
    }
    // 7) fused = attn + gate * au_hs -> f16
    fuse_gate<<<cgrid(M * D), 256, 0, stream>>>(attnO, gatef, auhf, fusedh, M * D);

    // 8) out = fused @ w_out + b_out + residual -> d_out (f32)
    {
        dim3 g(D / 64, M / 16);
        gemm_wmma<4><<<g, 32, 0, stream>>>(fusedh, D, woutt, D, D, out, nullptr, D, bout, hs);
    }
}
